// Deformation_45475113730114
// MI455X (gfx1250) — matrix operand order repaired and checked by reading, not compiled, not run on hardware
//
#include <hip/hip_runtime.h>

// ---------------------------------------------------------------------------
// Deformation field (HexPlane sample -> 10-layer MLP) for gfx1250 (MI455X).
// Compute-bound (~458 GFLOP vs ~200MB I/O) -> bf16 WMMA 16x16x32, f32 accum.
// ---------------------------------------------------------------------------

#define MT      32          // points per workgroup (2 M-tiles of 16)
#define MTILES  2
#define HPAD    264         // LDS row stride in halves (256 + 8 pad, 16B-aligned)

typedef __attribute__((ext_vector_type(16))) __bf16 v16bf;
typedef __attribute__((ext_vector_type(8)))  float  v8f;

struct alignas(16) U128 { unsigned int x, y, z, w; };
union Frag { v16bf v; U128 q[2]; };

__device__ __forceinline__ unsigned short f2bf(float f) {
    unsigned int u = __float_as_uint(f);
    u += 0x7FFFu + ((u >> 16) & 1u);          // round-to-nearest-even
    return (unsigned short)(u >> 16);
}

__device__ __forceinline__ int iclamp(int v, int lo, int hi) {
    return v < lo ? lo : (v > hi ? hi : v);
}

// grid_sample bilinear, align_corners=True, clamped indices (matches reference)
__device__ __forceinline__ float bilerp1(const float* __restrict__ plane,
                                         int W, int H, int c, float x, float y) {
    float fx = (x + 1.f) * 0.5f * (float)(W - 1);
    float fy = (y + 1.f) * 0.5f * (float)(H - 1);
    float x0 = floorf(fx), y0 = floorf(fy);
    float wx = fx - x0,   wy = fy - y0;
    int x0i = iclamp((int)x0, 0, W - 1);
    int x1i = x0i + 1 > W - 1 ? W - 1 : x0i + 1;
    int y0i = iclamp((int)y0, 0, H - 1);
    int y1i = y0i + 1 > H - 1 ? H - 1 : y0i + 1;
    const float* pc = plane + (size_t)c * (size_t)H * (size_t)W;
    float v00 = pc[(size_t)y0i * W + x0i];
    float v01 = pc[(size_t)y0i * W + x1i];
    float v10 = pc[(size_t)y1i * W + x0i];
    float v11 = pc[(size_t)y1i * W + x1i];
    return (v00 * (1.f - wx) + v01 * wx) * (1.f - wy) +
           (v10 * (1.f - wx) + v11 * wx) * wy;
}

// ---------------------------------------------------------------------------
// Weight / bias repacking.
// Packed B-fragment layout per (kt,nt) tile: 32 lanes x 16 halves contiguous.
//   lane L supplies column n = nt*16 + (L&15), K = kt*32 + (L>>4)*16 + e.
// Tile offsets (512 halves/tile): l0: 0..63 (4kt x 16nt),
//   l1..8: 64 + 128*(l-1) (8kt x 16nt), l9: 1088 (8kt x 4nt, cols padded to 64).
// ---------------------------------------------------------------------------
struct PrepArgs {
    const float* w[10];
    const float* b[10];
    unsigned short* wpk;
    float* bpk;
};

__global__ __launch_bounds__(256) void pack_weights_k(PrepArgs A) {
    const int TOTAL = 1120 * 512;
    int idx = blockIdx.x * 256 + threadIdx.x;
    if (idx >= TOTAL) return;
    int tile   = idx >> 9;
    int within = idx & 511;
    int lane   = within >> 4;
    int e      = within & 15;
    int l, lt;
    if (tile < 64)        { l = 0; lt = tile; }
    else if (tile < 1088) { l = 1 + (tile - 64) / 128; lt = (tile - 64) % 128; }
    else                  { l = 9; lt = tile - 1088; }
    int NT = (l == 9) ? 4 : 16;
    int kt = lt / NT, nt = lt % NT;
    int k  = kt * 32 + (lane >> 4) * 16 + e;
    int n  = nt * 16 + (lane & 15);
    int nsrc = (l == 9) ? 59 : 256;
    float v = (n < nsrc) ? A.w[l][(size_t)k * nsrc + n] : 0.f;
    A.wpk[idx] = f2bf(v);
}

__global__ __launch_bounds__(256) void pack_bias_k(PrepArgs A) {
    int idx = blockIdx.x * 256 + threadIdx.x;
    if (idx >= 10 * 256) return;
    int l = idx >> 8, n = idx & 255;
    int nsrc = (l == 9) ? 59 : 256;
    A.bpk[idx] = (n < nsrc) ? A.b[l][n] : 0.f;
}

// ---------------------------------------------------------------------------
// Main fused kernel
// ---------------------------------------------------------------------------
struct MlpArgs {
    const float* plane[24];
    const float* rays;
    const float* scl;
    const float* rot;
    const float* opa;
    const float* shs;
    const float* tim;
    const unsigned short* wpk;
    const float* bpk;
    float* out;
    int N;
};

// One GEMM layer: D(16x16 per tile) = A(hin) x B(wl) + bias, optional ReLU.
// Wave-uniform control flow only (EXEC all ones around WMMA).
template <int KT, int NT, bool FINAL>
__device__ __forceinline__ void mlp_layer(const unsigned short (*hin)[HPAD],
                                          unsigned short (*hout)[HPAD],
                                          float (*dout)[64],
                                          const unsigned short* __restrict__ wl,
                                          const float* __restrict__ bias,
                                          int lane, int wave) {
    const int mrow = lane & 15;
    const int hi   = lane >> 4;       // 0/1: which K-half this lane holds
    const int n    = lane & 15;
    for (int nt = wave; nt < NT; nt += 8) {
        v8f acc[MTILES] = {};
#pragma unroll
        for (int kt = 0; kt < KT; ++kt) {
            Frag b;
            const unsigned short* wp =
                wl + (((size_t)(kt * NT + nt)) * 32 + (size_t)lane) * 16;
            b.q[0] = *(const U128*)(wp);
            b.q[1] = *(const U128*)(wp + 8);
            const int kofs = kt * 32 + hi * 16;
#pragma unroll
            for (int mt = 0; mt < MTILES; ++mt) {
                Frag a;
                const unsigned short* r0 = &hin[mt * 16 + mrow][kofs];
                a.q[0] = *(const U128*)(r0);        // K = kofs .. kofs+7
                a.q[1] = *(const U128*)(r0 + 8);    // K = kofs+8 .. kofs+15
                acc[mt] = __builtin_amdgcn_wmma_f32_16x16x32_bf16(
                    false, a.v, false, b.v, (short)0, acc[mt], false, false);
            }
        }
        const float bv = bias[nt * 16 + n];
#pragma unroll
        for (int mt = 0; mt < MTILES; ++mt) {
#pragma unroll
            for (int r = 0; r < 8; ++r) {
                int m = mt * 16 + r + 8 * hi;       // C/D layout: M = r + 8*(lane/16)
                float v = acc[mt][r] + bv;
                if (FINAL) dout[m][nt * 16 + n] = v;
                else       hout[m][nt * 16 + n] = f2bf(fmaxf(v, 0.f));
            }
        }
    }
}

__global__ __launch_bounds__(256, 1) void deform_mlp_k(MlpArgs A) {
    __shared__ unsigned short hA[2][MT][HPAD];   // activation ping-pong (bf16)
    __shared__ float dOut[MT][64];               // final head output (f32)

    const int t    = threadIdx.x;
    const int lane = t & 31;
    const int wave = t >> 5;
    const int p0   = blockIdx.x * MT;

    const int CI[6] = {0, 0, 0, 1, 1, 2};
    const int CJ[6] = {1, 2, 3, 2, 3, 3};

    // ---- Phase 1: plane featurization -> hA[0][m][0..127] (bf16) ----
    for (int task = t; task < MT * 32; task += 256) {
        int m = task >> 5, c = task & 31;
        int p = p0 + m; if (p >= A.N) p = A.N - 1;
        float q[4];
        q[0] = -A.rays[(size_t)p * 3 + 0] * (1.f / 1.6f);
        q[1] = -A.rays[(size_t)p * 3 + 1] * (1.f / 1.6f);
        q[2] = -A.rays[(size_t)p * 3 + 2] * (1.f / 1.6f);
        q[3] =  A.tim[p];
#pragma unroll
        for (int ri = 0; ri < 4; ++ri) {
            int S = 64 << ri;
            float f = 1.f;
#pragma unroll
            for (int ci = 0; ci < 6; ++ci) {
                int j = CJ[ci];
                int W = S;
                int H = (j == 3) ? 75 : S;
                f *= bilerp1(A.plane[ri * 6 + ci], W, H, c, q[CI[ci]], q[j]);
            }
            hA[0][m][ri * 32 + c] = f2bf(f);
        }
    }
    __syncthreads();

    // ---- Phase 2: MLP ----
    // layer 0: 128 -> 256
    mlp_layer<4, 16, false>(hA[0], hA[1], dOut, A.wpk, A.bpk, lane, wave);
    __syncthreads();
    int cur = 1;
    // layers 1..7 (feat) + layer 8 (head0): 256 -> 256, ReLU
#pragma unroll 1
    for (int l = 1; l <= 8; ++l) {
        const unsigned short* wl = A.wpk + (size_t)(64 + 128 * (l - 1)) * 512;
        mlp_layer<8, 16, false>(hA[cur], hA[cur ^ 1], dOut, wl,
                                A.bpk + l * 256, lane, wave);
        __syncthreads();
        cur ^= 1;
    }
    // layer 9 (head1): 256 -> 59 (padded 64), no ReLU, f32 out
    mlp_layer<8, 4, true>(hA[cur], hA[cur ^ 1], dOut,
                          A.wpk + (size_t)1088 * 512, A.bpk + 9 * 256,
                          lane, wave);
    __syncthreads();

    // ---- Phase 3: residual compose into concatenated outputs ----
    float* outP = A.out;
    float* outS = A.out + (size_t)3 * A.N;
    float* outR = A.out + (size_t)6 * A.N;
    float* outO = A.out + (size_t)10 * A.N;
    float* outH = A.out + (size_t)11 * A.N;
    for (int idx = t; idx < MT * 59; idx += 256) {
        int m = idx / 59, o = idx % 59;
        int p = p0 + m;
        if (p >= A.N) continue;
        float d = dOut[m][o];
        if (o < 3)        outP[(size_t)p * 3 + o]        = A.rays[(size_t)p * 3 + o] + d;
        else if (o < 6)   outS[(size_t)p * 3 + (o - 3)]  = A.scl[(size_t)p * 3 + (o - 3)] + d;
        else if (o < 10)  outR[(size_t)p * 4 + (o - 6)]  = A.rot[(size_t)p * 4 + (o - 6)] + d;
        else if (o == 10) outO[p]                        = A.opa[p] + d;
        else              outH[(size_t)p * 48 + (o - 11)] =
                              A.shs[(size_t)p * 48 + (o - 11)] + d;
    }
}

// ---------------------------------------------------------------------------
// Host launcher
// ---------------------------------------------------------------------------
extern "C" void kernel_launch(void* const* d_in, const int* in_sizes, int n_in,
                              void* d_out, int out_size, void* d_ws, size_t ws_size,
                              hipStream_t stream) {
    const float* rays = (const float*)d_in[0];
    const float* scl  = (const float*)d_in[1];
    const float* rot  = (const float*)d_in[2];
    const float* opa  = (const float*)d_in[3];
    const float* shs  = (const float*)d_in[4];
    const float* tim  = (const float*)d_in[5];
    const int N = in_sizes[0] / 3;

    // Identify params by element count (robust to pytree flatten order).
    const float* planes[24] = {};
    const float* feat_w[8]  = {};
    const float* feat_b[8]  = {};
    const float* head_w[2]  = {};
    const float* head_b[2]  = {};
    int np = 0, n65536 = 0, n256 = 0;
    for (int i = 6; i < n_in; ++i) {
        int s = in_sizes[i];
        const float* p = (const float*)d_in[i];
        if (s == 32768)      { feat_w[0] = p; }
        else if (s == 65536) { if (n65536 < 7) feat_w[1 + n65536] = p;
                               else head_w[0] = p; ++n65536; }
        else if (s == 15104) { head_w[1] = p; }
        else if (s == 256)   { if (n256 < 8) feat_b[n256] = p;
                               else head_b[0] = p; ++n256; }
        else if (s == 59)    { head_b[1] = p; }
        else                 { if (np < 24) planes[np++] = p; }
    }

    // Workspace: packed bf16 weights (1120 tiles * 512 halves) + f32 biases.
    unsigned short* wpk = (unsigned short*)d_ws;
    float* bpk = (float*)((char*)d_ws + (size_t)1120 * 512 * 2);

    PrepArgs P;
    for (int l = 0; l < 8; ++l) { P.w[l] = feat_w[l]; P.b[l] = feat_b[l]; }
    P.w[8] = head_w[0]; P.b[8] = head_b[0];
    P.w[9] = head_w[1]; P.b[9] = head_b[1];
    P.wpk = wpk; P.bpk = bpk;

    pack_weights_k<<<(1120 * 512 + 255) / 256, 256, 0, stream>>>(P);
    pack_bias_k<<<10, 256, 0, stream>>>(P);

    MlpArgs M;
    for (int k = 0; k < 24; ++k) M.plane[k] = planes[k];
    M.rays = rays; M.scl = scl; M.rot = rot; M.opa = opa;
    M.shs = shs;   M.tim = tim;
    M.wpk = wpk;   M.bpk = bpk;
    M.out = (float*)d_out;
    M.N = N;

    deform_mlp_k<<<(N + MT - 1) / MT, 256, 0, stream>>>(M);
}
